// QuantLlamaMLP_33277406609703
// MI455X (gfx1250) — matrix operand order repaired and checked by reading, not compiled, not run on hardware
//
#include <hip/hip_runtime.h>
#include <stdint.h>

// ---- problem constants (from reference) ----
#define HIDDEN 4096
#define INTER  11008
#define MTOT   8192          // B*S = 4*2048
#define WGU    (2*INTER)     // 22016 gate|up columns

// ---- tiling ----
#define BM  128              // block rows (tokens)
#define BN  64               // block cols
#define KT  32               // WMMA K step (f16)
#define LDA 40               // padded LDS row stride in halfs (80B, 16B-aligned)

typedef __attribute__((ext_vector_type(16))) _Float16 v16h;
typedef __attribute__((ext_vector_type(8)))  _Float16 v8h;
typedef __attribute__((ext_vector_type(4)))  _Float16 v4h;
typedef __attribute__((ext_vector_type(8)))  float    v8f;
typedef int v4i __attribute__((vector_size(16)));   // matches builtin param type

// ---------- CDNA5 async global->LDS copy (ASYNCcnt path), guarded ----------
#if defined(__AMDGCN__) && __has_builtin(__builtin_amdgcn_global_load_async_to_lds_b128)
#define ASYNC_LDS 1
#else
#define ASYNC_LDS 0
#endif

__device__ __forceinline__ void cp16(void* lds, const void* g) {
#if ASYNC_LDS
    // (global int4*, lds int4*, imm offset, imm cpol)
    __builtin_amdgcn_global_load_async_to_lds_b128(
        (__attribute__((address_space(1))) v4i*)(uintptr_t)g,
        (__attribute__((address_space(3))) v4i*)(uint32_t)(uintptr_t)lds,
        /*offset=*/0, /*cpol=*/0);
#else
    *reinterpret_cast<uint4*>(lds) = *reinterpret_cast<const uint4*>(g);
#endif
}

__device__ __forceinline__ void wait_async() {
#if ASYNC_LDS
#if __has_builtin(__builtin_amdgcn_s_wait_asynccnt)
    __builtin_amdgcn_s_wait_asynccnt(0);
#else
    asm volatile("s_wait_asynccnt 0x0" ::: "memory");
#endif
#endif
}

__device__ __forceinline__ v16h cat16(v8h lo, v8h hi) {
    v16h r;
#pragma unroll
    for (int i = 0; i < 8; ++i) { r[i] = lo[i]; r[i + 8] = hi[i]; }
    return r;
}

__device__ __forceinline__ v8f wmma_f16(v16h a, v16h b, v8f c) {
    // D = A(16x32 f16) x B(32x16 f16) + C(16x16 f32)
    return __builtin_amdgcn_wmma_f32_16x16x32_f16(
        /*neg_a=*/false, a, /*neg_b=*/false, b,
        /*c_mod=*/(short)0, c, /*reuse_a=*/false, /*reuse_b=*/false);
}

// Dequant 8 consecutive k-nibbles of one column into f16: w = s*nib - s*(z+1)
__device__ __forceinline__ v8h dq8(uint32_t w, float s, uint32_t z) {
    const float b = -s * (float)(z + 1u);
    v8h o;
#pragma unroll
    for (int i = 0; i < 8; ++i)
        o[i] = (_Float16)(s * (float)((w >> (4 * i)) & 15u) + b);
    return o;
}

// ---------------- kernel 0: x fp32 -> f16 ----------------
__global__ __launch_bounds__(256)
void cvt_f32_f16(const float* __restrict__ x, _Float16* __restrict__ y) {
    const size_t i = ((size_t)blockIdx.x * blockDim.x + threadIdx.x) * 4;
    const float4 v = *reinterpret_cast<const float4*>(x + i);
    v4h o = { (_Float16)v.x, (_Float16)v.y, (_Float16)v.z, (_Float16)v.w };
    *reinterpret_cast<v4h*>(y + i) = o;
}

// ---- staging helpers (issue copies / dequant stores; no barriers inside) ----
__device__ __forceinline__ void stage_gu(int tid, int k0, int m0, int n0,
                                         _Float16* As, _Float16* Bg, _Float16* Bu,
                                         const _Float16* __restrict__ xh,
                                         const uint32_t* __restrict__ qw,
                                         const uint32_t* __restrict__ qz,
                                         const float*    __restrict__ sc) {
    // A tile (BM x KT): pure copy -> async path
#pragma unroll
    for (int j = 0; j < 2; ++j) {
        const int c   = tid + 256 * j;      // 512 chunks of 16B
        const int row = c >> 2;
        const int ko  = (c & 3) * 8;
        cp16(&As[row * LDA + ko], xh + (size_t)(m0 + row) * HIDDEN + k0 + ko);
    }
    // B tiles (KT x BN, stored [n][k]): dequant on VALU, ds_store
    const int dq_n  = tid & 63;
    const int dq_r  = tid >> 6;             // 0..3
    const int col_g = n0 + dq_n;
    const int col_u = col_g + INTER;
    const int g     = k0 >> 7;              // k0 / GROUP
    const size_t wrow = (size_t)((k0 >> 3) + dq_r) * WGU;
    {
        const uint32_t w = qw[wrow + col_g];
        const uint32_t z = (qz[(size_t)g * (WGU >> 3) + (col_g >> 3)] >> ((col_g & 7) * 4)) & 15u;
        const float    s = sc[(size_t)g * WGU + col_g];
        *reinterpret_cast<v8h*>(&Bg[dq_n * LDA + dq_r * 8]) = dq8(w, s, z);
    }
    {
        const uint32_t w = qw[wrow + col_u];
        const uint32_t z = (qz[(size_t)g * (WGU >> 3) + (col_u >> 3)] >> ((col_u & 7) * 4)) & 15u;
        const float    s = sc[(size_t)g * WGU + col_u];
        *reinterpret_cast<v8h*>(&Bu[dq_n * LDA + dq_r * 8]) = dq8(w, s, z);
    }
    // prefetch the K-step after next (speculative, no counter)
    __builtin_prefetch(qw + wrow + 2 * (KT / 8) * WGU + col_g, 0, 1);
}

__device__ __forceinline__ void stage_dn(int tid, int k0, int m0, int n0,
                                         _Float16* As, _Float16* Bs,
                                         const _Float16* __restrict__ hbuf,
                                         const uint32_t* __restrict__ qw,
                                         const uint32_t* __restrict__ qz,
                                         const float*    __restrict__ sc) {
#pragma unroll
    for (int j = 0; j < 2; ++j) {
        const int c   = tid + 256 * j;
        const int row = c >> 2;
        const int ko  = (c & 3) * 8;
        cp16(&As[row * LDA + ko], hbuf + (size_t)(m0 + row) * INTER + k0 + ko);
    }
    const int dq_n = tid & 63;
    const int dq_r = tid >> 6;
    const int col  = n0 + dq_n;
    const int g    = k0 >> 7;
    const size_t wrow = (size_t)((k0 >> 3) + dq_r) * HIDDEN;
    const uint32_t w = qw[wrow + col];
    const uint32_t z = (qz[(size_t)g * (HIDDEN >> 3) + (col >> 3)] >> ((col & 7) * 4)) & 15u;
    const float    s = sc[(size_t)g * HIDDEN + col];
    *reinterpret_cast<v8h*>(&Bs[dq_n * LDA + dq_r * 8]) = dq8(w, s, z);
    __builtin_prefetch(qw + wrow + 2 * (KT / 8) * HIDDEN + col, 0, 1);
}

// ---------------- kernel 1: gate/up GEMM + SiLU fused ----------------
// grid = (INTER/BN, MTOT/BM), block = 256 (8 waves, 4(M) x 2(N)); double-buffered LDS
__global__ __launch_bounds__(256)
void gateup_kernel(const _Float16* __restrict__ xh,
                   const uint32_t* __restrict__ qw,
                   const uint32_t* __restrict__ qz,
                   const float*    __restrict__ sc,
                   _Float16*       __restrict__ hbuf) {
    __shared__ __align__(16) _Float16 As[2][BM * LDA];
    __shared__ __align__(16) _Float16 Bg[2][BN * LDA];
    __shared__ __align__(16) _Float16 Bu[2][BN * LDA];

    const int n0   = blockIdx.x * BN;   // h-column tile (gate col; up col = +INTER)
    const int m0   = blockIdx.y * BM;
    const int tid  = threadIdx.x;
    const int lane = tid & 31;
    const int wave = tid >> 5;
    const int wm   = wave & 3;          // 4 waves x 32 rows = 128
    const int wn   = wave >> 2;         // 2 waves x 32 cols = 64
    const int hl   = lane >> 4;         // lane half (0/1)
    const int l16  = lane & 15;

    v8f accg[2][2] = {};
    v8f accu[2][2] = {};

    const int nkt = HIDDEN / KT;        // 128

    // prologue: stage step 0 into buffer 0
    stage_gu(tid, 0, m0, n0, As[0], Bg[0], Bu[0], xh, qw, qz, sc);
    wait_async();
    __syncthreads();

    for (int kt = 0; kt < nkt; ++kt) {
        const int cur = kt & 1;
        // issue next stage (async A copies + dequant B) into the other buffer
        if (kt + 1 < nkt)
            stage_gu(tid, (kt + 1) * KT, m0, n0, As[cur ^ 1], Bg[cur ^ 1], Bu[cur ^ 1],
                     xh, qw, qz, sc);

        // ---- fragments (contiguous k = 16*h + s) + 8 WMMAs per wave ----
        const _Float16* a = As[cur];
        const _Float16* bg = Bg[cur];
        const _Float16* bu = Bu[cur];
        v16h afr[2], bgf[2], buf_[2];
#pragma unroll
        for (int ti = 0; ti < 2; ++ti) {
            const int row = wm * 32 + ti * 16 + l16;
            afr[ti] = cat16(*reinterpret_cast<const v8h*>(&a[row * LDA + hl * 16]),
                            *reinterpret_cast<const v8h*>(&a[row * LDA + hl * 16 + 8]));
        }
#pragma unroll
        for (int tj = 0; tj < 2; ++tj) {
            const int col = wn * 32 + tj * 16 + l16;
            const int ko  = hl * 16;
            bgf[tj]  = cat16(*reinterpret_cast<const v8h*>(&bg[col * LDA + ko]),
                             *reinterpret_cast<const v8h*>(&bg[col * LDA + ko + 8]));
            buf_[tj] = cat16(*reinterpret_cast<const v8h*>(&bu[col * LDA + ko]),
                             *reinterpret_cast<const v8h*>(&bu[col * LDA + ko + 8]));
        }
#pragma unroll
        for (int ti = 0; ti < 2; ++ti)
#pragma unroll
            for (int tj = 0; tj < 2; ++tj) {
                accg[ti][tj] = wmma_f16(afr[ti], bgf[tj],  accg[ti][tj]);
                accu[ti][tj] = wmma_f16(afr[ti], buf_[tj], accu[ti][tj]);
            }

        // next buffer fully resident (ASYNCcnt + DScnt) before anyone reads it
        wait_async();
        __syncthreads();
    }

    // ---- epilogue: h = silu(gate) * up, store f16 ----
#pragma unroll
    for (int ti = 0; ti < 2; ++ti)
#pragma unroll
        for (int tj = 0; tj < 2; ++tj)
#pragma unroll
            for (int r = 0; r < 8; ++r) {
                const float gv = accg[ti][tj][r];
                const float uv = accu[ti][tj][r];
                const float hv = (gv / (1.0f + __expf(-gv))) * uv;
                const int m = m0 + wm * 32 + ti * 16 + hl * 8 + r;
                const int n = n0 + wn * 32 + tj * 16 + l16;
                hbuf[(size_t)m * INTER + n] = (_Float16)hv;
            }
}

// ---------------- kernel 2: down GEMM ----------------
// grid = (HIDDEN/BN, MTOT/BM), block = 256; double-buffered LDS
__global__ __launch_bounds__(256)
void down_kernel(const _Float16* __restrict__ hbuf,
                 const uint32_t* __restrict__ qw,
                 const uint32_t* __restrict__ qz,
                 const float*    __restrict__ sc,
                 float*          __restrict__ out) {
    __shared__ __align__(16) _Float16 As[2][BM * LDA];
    __shared__ __align__(16) _Float16 Bs[2][BN * LDA];

    const int n0   = blockIdx.x * BN;
    const int m0   = blockIdx.y * BM;
    const int tid  = threadIdx.x;
    const int lane = tid & 31;
    const int wave = tid >> 5;
    const int wm   = wave & 3;
    const int wn   = wave >> 2;
    const int hl   = lane >> 4;
    const int l16  = lane & 15;

    v8f acc[2][2] = {};

    const int nkt = INTER / KT;         // 344

    stage_dn(tid, 0, m0, n0, As[0], Bs[0], hbuf, qw, qz, sc);
    wait_async();
    __syncthreads();

    for (int kt = 0; kt < nkt; ++kt) {
        const int cur = kt & 1;
        if (kt + 1 < nkt)
            stage_dn(tid, (kt + 1) * KT, m0, n0, As[cur ^ 1], Bs[cur ^ 1], hbuf, qw, qz, sc);

        const _Float16* a = As[cur];
        const _Float16* b = Bs[cur];
        v16h afr[2], bfr[2];
#pragma unroll
        for (int ti = 0; ti < 2; ++ti) {
            const int row = wm * 32 + ti * 16 + l16;
            afr[ti] = cat16(*reinterpret_cast<const v8h*>(&a[row * LDA + hl * 16]),
                            *reinterpret_cast<const v8h*>(&a[row * LDA + hl * 16 + 8]));
        }
#pragma unroll
        for (int tj = 0; tj < 2; ++tj) {
            const int c  = wn * 32 + tj * 16 + l16;
            const int ko = hl * 16;
            bfr[tj] = cat16(*reinterpret_cast<const v8h*>(&b[c * LDA + ko]),
                            *reinterpret_cast<const v8h*>(&b[c * LDA + ko + 8]));
        }
#pragma unroll
        for (int ti = 0; ti < 2; ++ti)
#pragma unroll
            for (int tj = 0; tj < 2; ++tj)
                acc[ti][tj] = wmma_f16(afr[ti], bfr[tj], acc[ti][tj]);

        wait_async();
        __syncthreads();
    }

#pragma unroll
    for (int ti = 0; ti < 2; ++ti)
#pragma unroll
        for (int tj = 0; tj < 2; ++tj)
#pragma unroll
            for (int r = 0; r < 8; ++r) {
                const int m = m0 + wm * 32 + ti * 16 + hl * 8 + r;
                const int n = n0 + wn * 32 + tj * 16 + l16;
                out[(size_t)m * HIDDEN + n] = acc[ti][tj][r];
            }
}

extern "C" void kernel_launch(void* const* d_in, const int* in_sizes, int n_in,
                              void* d_out, int out_size, void* d_ws, size_t ws_size,
                              hipStream_t stream) {
    (void)in_sizes; (void)n_in; (void)out_size; (void)ws_size;
    const float*    x     = (const float*)   d_in[0];
    const uint32_t* qw_gu = (const uint32_t*)d_in[1];
    const uint32_t* qz_gu = (const uint32_t*)d_in[2];
    const float*    sc_gu = (const float*)   d_in[3];
    // d_in[4] = g_idx_gu (arange/GROUP, computed analytically)
    const uint32_t* qw_d  = (const uint32_t*)d_in[5];
    const uint32_t* qz_d  = (const uint32_t*)d_in[6];
    const float*    sc_d  = (const float*)   d_in[7];
    // d_in[8] = g_idx_down (analytic)

    char* ws = (char*)d_ws;
    _Float16* xh = (_Float16*)ws;                                   // 8192*4096*2  = 64 MB
    _Float16* hb = (_Float16*)(ws + (size_t)MTOT * HIDDEN * 2);     // 8192*11008*2 = 172 MB

    // 1) x -> f16
    cvt_f32_f16<<<(MTOT * HIDDEN) / (256 * 4), 256, 0, stream>>>(x, xh);
    // 2) gate/up GEMM + silu -> h (f16)
    gateup_kernel<<<dim3(INTER / BN, MTOT / BM), 256, 0, stream>>>(xh, qw_gu, qz_gu, sc_gu, hb);
    // 3) down GEMM -> out (fp32)
    down_kernel<<<dim3(HIDDEN / BN, MTOT / BM), 256, 0, stream>>>(hb, qw_d, qz_d, sc_d, (float*)d_out);
}
